// Linear3Bit_335007449962
// MI455X (gfx1250) — matrix operand order repaired and checked by reading, not compiled, not run on hardware
//
#include <hip/hip_runtime.h>

typedef _Float16 v8h  __attribute__((ext_vector_type(8)));
typedef _Float16 v16h __attribute__((ext_vector_type(16)));
typedef float    v8f  __attribute__((ext_vector_type(8)));

#define IN_F   4096
#define OUT_F  4096
#define M_TOT  8192
#define NGROUP 524288

#define BM 256
#define BN 128
#define KT 64
#define NSTAGE (IN_F / KT)
#define LDSS 80   // padded LDS row stride in halfs (160B: 16B aligned, bank-spread)

#if defined(__has_builtin)
#if __has_builtin(__builtin_amdgcn_global_load_async_to_lds_b128)
#define HAVE_ASYNC_LDS 1
#endif
#endif
#ifndef HAVE_ASYNC_LDS
#define HAVE_ASYNC_LDS 0
#endif

#if HAVE_ASYNC_LDS
// Builtin signature (from clang diagnostic): (AS1 int4*, AS3 int4*, imm offset, imm cpol)
typedef int ai4 __attribute__((vector_size(16)));
typedef __attribute__((address_space(1))) ai4 gas_i4;
typedef __attribute__((address_space(3))) ai4 las_i4;
__device__ __forceinline__ void async_copy16(const _Float16* g, _Float16* l) {
    __builtin_amdgcn_global_load_async_to_lds_b128((gas_i4*)g, (las_i4*)l, 0, 0);
}
__device__ __forceinline__ void async_wait0() {
#if __has_builtin(__builtin_amdgcn_s_wait_asynccnt)
    __builtin_amdgcn_s_wait_asynccnt(0);
#else
    asm volatile("s_wait_asynccnt 0" ::: "memory");
#endif
}
#endif

// ---------------------------------------------------------------------------
// Kernel 1: 3-bit dequant -> f16 W[N][K] (one group of 32 weights per thread)
// ---------------------------------------------------------------------------
__global__ __launch_bounds__(256) void dequant3_kernel(const int* __restrict__ q3,
                                                       const _Float16* __restrict__ wnrm,
                                                       _Float16* __restrict__ Wb) {
    int g = blockIdx.x * 256 + threadIdx.x;          // group id < 524288
    const int4* p4 = (const int4*)(q3 + (size_t)g * 12);
    int4 w0 = p4[0], w1 = p4[1], w2 = p4[2];
    int b[12] = {w0.x, w0.y, w0.z, w0.w, w1.x, w1.y, w1.z, w1.w, w2.x, w2.y, w2.z, w2.w};

    float n = (float)wnrm[g];
    float s = (2.0f * n + 1e-8f) * (1.0f / 7.0f);

    _Float16 out[32];
#pragma unroll
    for (int c = 0; c < 4; ++c) {
        int p0 = b[3 * c + 0] & 255;
        int p1 = b[3 * c + 1] & 255;
        int p2 = b[3 * c + 2] & 255;
        int q[8];
        q[0] = p0 & 7;
        q[1] = (p0 >> 3) & 7;
        q[2] = ((p0 >> 6) & 3) | ((p1 & 1) << 2);
        q[3] = (p1 >> 1) & 7;
        q[4] = (p1 >> 4) & 7;
        q[5] = ((p1 >> 7) & 1) | ((p2 & 3) << 1);
        q[6] = (p2 >> 2) & 7;
        q[7] = (p2 >> 5) & 7;
#pragma unroll
        for (int j = 0; j < 8; ++j)
            out[c * 8 + j] = (_Float16)((float)q[j] * s - n);
    }
    v8h* dst = (v8h*)(Wb + (size_t)g * 32);          // group is contiguous along K
#pragma unroll
    for (int i = 0; i < 4; ++i) dst[i] = *(const v8h*)(out + i * 8);
}

// ---------------------------------------------------------------------------
// Kernel 2: x f32 -> f16 (8 elements / thread)
// ---------------------------------------------------------------------------
__global__ __launch_bounds__(256) void xcvt_kernel(const float* __restrict__ x,
                                                   _Float16* __restrict__ Xb) {
    size_t i = ((size_t)blockIdx.x * 256 + threadIdx.x) * 8;
    const float4* s = (const float4*)(x + i);
    float4 f0 = s[0], f1 = s[1];
    v8h h;
    h[0] = (_Float16)f0.x; h[1] = (_Float16)f0.y;
    h[2] = (_Float16)f0.z; h[3] = (_Float16)f0.w;
    h[4] = (_Float16)f1.x; h[5] = (_Float16)f1.y;
    h[6] = (_Float16)f1.z; h[7] = (_Float16)f1.w;
    *(v8h*)(Xb + i) = h;
}

// ---------------------------------------------------------------------------
// Kernel 3: WMMA GEMM  out[M][N] = Xb[M][K] * Wb[N][K]^T + bias
// 256 threads = 8 wave32 arranged 4(M) x 2(N); block tile 256x128;
// wave tile 64x64 = 4x4 grid of 16x16 WMMA accumulators.
// Double-buffered LDS: async DMA for stage s+1 overlaps WMMA of stage s.
// ---------------------------------------------------------------------------
__global__ __launch_bounds__(256) void gemm_f16_wmma_kernel(const _Float16* __restrict__ Xb,
                                                            const _Float16* __restrict__ Wb,
                                                            const float* __restrict__ bias,
                                                            float* __restrict__ out) {
    __shared__ _Float16 sA[2][BM * LDSS];   // 2 x 40 KB
    __shared__ _Float16 sB[2][BN * LDSS];   // 2 x 20 KB

    const int tid     = threadIdx.x;
    const int lane    = tid & 31;
    const int wave    = tid >> 5;     // 0..7
    const int wm      = wave & 3;     // 4 waves along M, 64 rows each
    const int wn      = wave >> 2;    // 2 waves along N, 64 cols each
    const int laneLow = lane & 15;
    const int laneHi  = lane >> 4;    // 0 or 1

    const int m0 = blockIdx.y * BM;
    const int n0 = blockIdx.x * BN;

    // Issue the global->LDS copies for one K-stage into buffer `buf`.
    auto stage = [&](int buf, int kk) {
#pragma unroll
        for (int i = 0; i < 8; ++i) {             // A: 256 rows x 8 chunks of 16B
            int c   = tid + i * 256;
            int row = c >> 3;
            int ck  = (c & 7) * 8;
            const _Float16* g = &Xb[(size_t)(m0 + row) * IN_F + kk + ck];
            _Float16*       l = &sA[buf][row * LDSS + ck];
#if HAVE_ASYNC_LDS
            async_copy16(g, l);
#else
            *(v8h*)l = *(const v8h*)g;
#endif
        }
#pragma unroll
        for (int i = 0; i < 4; ++i) {             // B: 128 rows x 8 chunks of 16B
            int c   = tid + i * 256;
            int row = c >> 3;
            int ck  = (c & 7) * 8;
            const _Float16* g = &Wb[(size_t)(n0 + row) * IN_F + kk + ck];
            _Float16*       l = &sB[buf][row * LDSS + ck];
#if HAVE_ASYNC_LDS
            async_copy16(g, l);
#else
            *(v8h*)l = *(const v8h*)g;
#endif
        }
    };

    v8f acc[4][4];
#pragma unroll
    for (int mi = 0; mi < 4; ++mi)
#pragma unroll
        for (int ni = 0; ni < 4; ++ni)
#pragma unroll
            for (int r = 0; r < 8; ++r) acc[mi][ni][r] = 0.0f;

    stage(0, 0);   // prologue: stage 0 in flight

    for (int s = 0; s < NSTAGE; ++s) {
        const int buf = s & 1;
#if HAVE_ASYNC_LDS
        async_wait0();       // my loads for buf `buf` complete
#endif
        __syncthreads();     // everyone's loads visible; prev-stage buffer now free
        if (s + 1 < NSTAGE)  // overlap next stage DMA with this stage's WMMAs
            stage(buf ^ 1, (s + 1) * KT);

        const _Float16* pA = sA[buf];
        const _Float16* pB = sB[buf];
#pragma unroll
        for (int ks = 0; ks < 2; ++ks) {      // two K=32 WMMA steps per stage
            v16h af[4];
            v16h bf[4];
#pragma unroll
            for (int mi = 0; mi < 4; ++mi) {
                // A 16x32 f16 layout: lanes 0-15 row M=lane, K {0..7,16..23};
                // lanes 16-31 same rows, K {8..15,24..31}
                const _Float16* base =
                    &pA[(wm * 64 + mi * 16 + laneLow) * LDSS + ks * 32 + laneHi * 8];
                v8h lo = *(const v8h*)base;
                v8h hi = *(const v8h*)(base + 16);
                af[mi] = __builtin_shufflevector(lo, hi, 0, 1, 2, 3, 4, 5, 6, 7,
                                                 8, 9, 10, 11, 12, 13, 14, 15);
            }
#pragma unroll
            for (int ni = 0; ni < 4; ++ni) {
                // B 32x16 f16 layout: lane N=lane&15 holds K {0..7,16..23} (lanes<16) / {8..15,24..31}
                const _Float16* base =
                    &pB[(wn * 64 + ni * 16 + laneLow) * LDSS + ks * 32 + laneHi * 8];
                v8h lo = *(const v8h*)base;
                v8h hi = *(const v8h*)(base + 16);
                bf[ni] = __builtin_shufflevector(lo, hi, 0, 1, 2, 3, 4, 5, 6, 7,
                                                 8, 9, 10, 11, 12, 13, 14, 15);
            }
#pragma unroll
            for (int mi = 0; mi < 4; ++mi)
#pragma unroll
                for (int ni = 0; ni < 4; ++ni)
                    acc[mi][ni] = __builtin_amdgcn_wmma_f32_16x16x32_f16(
                        false, af[mi], false, bf[ni], (short)0, acc[mi][ni],
                        false, false);
        }
    }

    // Epilogue: C/D layout — VGPR r: M=r (lanes 0-15) / M=r+8 (lanes 16-31), N=lane&15
    // Non-temporal stores: output is write-once; keep Xb/Wb resident in L2.
#pragma unroll
    for (int mi = 0; mi < 4; ++mi) {
#pragma unroll
        for (int ni = 0; ni < 4; ++ni) {
            int   nn    = n0 + wn * 64 + ni * 16 + laneLow;
            float bv    = bias[nn];
            int   mbase = m0 + wm * 64 + mi * 16 + laneHi * 8;
#pragma unroll
            for (int r = 0; r < 8; ++r)
                __builtin_nontemporal_store(acc[mi][ni][r] + bv,
                                            &out[(size_t)(mbase + r) * OUT_F + nn]);
        }
    }
}

// ---------------------------------------------------------------------------
extern "C" void kernel_launch(void* const* d_in, const int* in_sizes, int n_in,
                              void* d_out, int out_size, void* d_ws, size_t ws_size,
                              hipStream_t stream) {
    const float*    x    = (const float*)d_in[0];
    const int*      q3   = (const int*)d_in[1];
    const _Float16* wnrm = (const _Float16*)d_in[2];
    const float*    bias = (const float*)d_in[3];
    float*          out  = (float*)d_out;

    _Float16* Wb = (_Float16*)d_ws;                                      // 32 MB
    _Float16* Xb = (_Float16*)((char*)d_ws +
                               (size_t)OUT_F * IN_F * sizeof(_Float16)); // +32 MB

    dequant3_kernel<<<NGROUP / 256, 256, 0, stream>>>(q3, wnrm, Wb);
    xcvt_kernel<<<((size_t)M_TOT * IN_F / 8) / 256, 256, 0, stream>>>(x, Xb);
    gemm_f16_wmma_kernel<<<dim3(OUT_F / BN, M_TOT / BM), 256, 0, stream>>>(Xb, Wb, bias, out);
}
